// BahdanauAttention_80951543595394
// MI455X (gfx1250) — matrix operand order repaired and checked by reading, not compiled, not run on hardware
//
#include <hip/hip_runtime.h>

typedef __attribute__((ext_vector_type(16))) __bf16 v16bf;
typedef __attribute__((ext_vector_type(8)))  float  v8f;

#if defined(__AMDGCN__) && __has_builtin(__builtin_amdgcn_tanhf)
#define DEV_TANH(x) __builtin_amdgcn_tanhf(x)
#else
#define DEV_TANH(x) tanhf(x)
#endif

// pointer types matching the async-to-LDS builtin's signature:
// param0: int __vector_size__(16) addrspace(1)*  (printed as "__device__" in HIP)
// param1: int __vector_size__(16) addrspace(3)*  (LDS / "__shared__")
typedef int v4i_vs __attribute__((vector_size(16)));
typedef __attribute__((address_space(1))) v4i_vs gv4i;
typedef __attribute__((address_space(3))) v4i_vs lv4i;

__device__ __forceinline__ void async_copy_b128(const void* gsrc, void* ldst) {
#if defined(__AMDGCN__) && __has_builtin(__builtin_amdgcn_global_load_async_to_lds_b128)
    __builtin_amdgcn_global_load_async_to_lds_b128(
        (gv4i*)gsrc, (lv4i*)ldst, /*offset=*/0, /*cpol=*/0);
#else
    *(uint4*)ldst = *(const uint4*)gsrc;   // synchronous fallback
#endif
}

__device__ __forceinline__ void wait_async_lds() {
#if defined(__AMDGCN__) && __has_builtin(__builtin_amdgcn_s_wait_asynccnt)
    __builtin_amdgcn_s_wait_asynccnt(0);
#elif defined(__AMDGCN__)
    asm volatile("s_wait_asynccnt 0x0" ::: "memory");
#endif
}

__device__ __forceinline__ unsigned short f32_to_bf16(float f) {
    union { float f; unsigned int u; } x{f};
    unsigned int r = x.u + 0x7FFFu + ((x.u >> 16) & 1u);   // round-to-nearest-even
    return (unsigned short)(r >> 16);
}

union Frag { v16bf v; unsigned int u[8]; };

// ---------------------------------------------------------------------------
// Elementwise f32 -> bf16 conversion
// ---------------------------------------------------------------------------
__global__ __launch_bounds__(256) void cvt_bf16_kernel(
    const float* __restrict__ in, unsigned short* __restrict__ out, int n) {
    int i = blockIdx.x * blockDim.x + threadIdx.x;
    if (i < n) out[i] = f32_to_bf16(in[i]);
}

// ---------------------------------------------------------------------------
// Batched GEMM: C[bz] = A[bz] (MxK bf16) * B[bz] (KxN bf16) + bias  (f32 out)
//
// Block tile 64x128, 256 threads = 8 waves.  Wave (wm = wave&3, nh = wave>>2)
// owns row-tile wm and four col-tiles nh*4..nh*4+3  -> 4 WMMAs per K-step,
// with one A fragment reused across all four (register blocking).
// K stepped by 32.  LDS double-buffered:
//   - A tile (64x32) staged with global_load_async_to_lds_b128 (ASYNCcnt)
//   - B tile (32x128) staged coalesced b128 + transposed into LDS so each
//     fragment element is one packed-dword DS read
// LDS row stride = 40 u16 (80 B): keeps b128 stores 16B-aligned and makes the
// 16-lane row-strided fragment reads bank-conflict-free (20*m mod 64 distinct).
// Requires: M%64==0, N%128==0, K%32==0 (true for all three GEMMs here).
// ---------------------------------------------------------------------------
__global__ __launch_bounds__(256) void gemm_bf16_wmma(
    const unsigned short* __restrict__ A, const unsigned short* __restrict__ B,
    const float* __restrict__ bias, float* __restrict__ C,
    int M, int N, int K, long strideA, long strideB, long strideC) {
    constexpr int LDA = 40;                       // u16 stride inside LDS tiles
    __shared__ __align__(16) unsigned short sA[2][64 * LDA];
    __shared__ __align__(16) unsigned short sB[2][128 * LDA];  // transposed [n][k]

    const int tid  = threadIdx.x;
    const int lane = tid & 31;
    const int wave = tid >> 5;
    const int m16  = lane & 15;
    const int h    = lane >> 4;

    const int m0 = blockIdx.y * 64;
    const int n0 = blockIdx.x * 128;
    const unsigned short* Ab = A + (long)blockIdx.z * strideA;
    const unsigned short* Bb = B + (long)blockIdx.z * strideB;
    float*                Cb = C + (long)blockIdx.z * strideC;

    const int wm  = wave & 3;                    // row-tile 0..3
    const int wnb = (wave >> 2) * 4;             // first of four col-tiles

    // staging decomposition
    const int arow = tid >> 2;                   // 0..63   (A row)
    const int aoff = (tid & 3) * 8;              // 0/8/16/24 u16 (A k-offset)
    const int brow = tid >> 3;                   // 0..31   (B k-row)
    const int boff = (tid & 7) * 16;             // 0..112  (B n-offset)

    auto stage = [&](int buf, int k0) {
        // --- A tile: 64x32 u16, one async b128 per thread ---
        const unsigned short* gA = Ab + (long)(m0 + arow) * K + k0 + aoff;
        async_copy_b128(gA, &sA[buf][arow * LDA + aoff]);
        // --- B tile: 32x128 u16, two coalesced b128 loads, transpose to LDS ---
        const unsigned short* gB = Bb + (long)(k0 + brow) * N + n0 + boff;
        uint4 v0 = *(const uint4*)gB;
        uint4 v1 = *(const uint4*)(gB + 8);
        const unsigned short* e0 = (const unsigned short*)&v0;
        const unsigned short* e1 = (const unsigned short*)&v1;
#pragma unroll
        for (int j = 0; j < 8; ++j) {
            sB[buf][(boff + j)     * LDA + brow] = e0[j];
            sB[buf][(boff + 8 + j) * LDA + brow] = e1[j];
        }
    };

    v8f acc[4] = {};
    stage(0, 0);
    wait_async_lds();
    __syncthreads();

    const int ksteps = K >> 5;
    for (int ks = 0; ks < ksteps; ++ks) {
        const int cur = ks & 1;
        if (ks + 1 < ksteps) stage(cur ^ 1, (ks + 1) * 32);   // overlap next tile

        // A fragment (16x32): VGPR v<4: K=2v+8h ; v>=4: K=16+2(v-4)+8h
        Frag a;
#pragma unroll
        for (int v = 0; v < 8; ++v) {
            int kk = ((v < 4) ? (v * 2) : (16 + (v - 4) * 2)) + h * 8;
            a.u[v] = *(const unsigned int*)&sA[cur][(wm * 16 + m16) * LDA + kk];
        }
        // four B fragments + four WMMAs, A reused
#pragma unroll
        for (int t = 0; t < 4; ++t) {
            Frag b;
            const int n = (wnb + t) * 16 + m16;
#pragma unroll
            for (int v = 0; v < 8; ++v)
                b.u[v] = *(const unsigned int*)&sB[cur][n * LDA + ((v < 4) ? (v * 2) : (16 + (v - 4) * 2)) + h * 8];
            acc[t] = __builtin_amdgcn_wmma_f32_16x16x32_bf16(
                false, a.v, false, b.v, (short)0, acc[t], false, false);
        }
        wait_async_lds();   // next tile's async copies done before barrier
        __syncthreads();
    }

    // C/D layout: VGPR r -> row (wm*16 + r + 8h), col = tile*16 + lane%16
#pragma unroll
    for (int t = 0; t < 4; ++t) {
        const int col = n0 + (wnb + t) * 16 + m16;
        const float bc = bias ? bias[col] : 0.0f;
#pragma unroll
        for (int r = 0; r < 8; ++r) {
            int row = m0 + wm * 16 + r + 8 * h;
            Cb[(long)row * N + col] = acc[t][r] + bc;
        }
    }
}

// ---------------------------------------------------------------------------
// Score + mask + softmax.  One 256-thread block (8 waves) per (b,t) row.
// Wave w handles s = w, w+8, ...; lanes read float4 (global_load_b128,
// coalesced), hardware v_tanh_f32, wave32 butterfly reduction.
// ---------------------------------------------------------------------------
__global__ __launch_bounds__(256) void score_softmax_kernel(
    const float* __restrict__ q,              // [B,T,D]
    const float* __restrict__ k,              // [B,S,D]
    const float* __restrict__ scale,          // [D]
    const unsigned char* __restrict__ mask,   // [B,S] bool
    float* __restrict__ attn_out,             // [B,T,S] f32 (output #2)
    unsigned short* __restrict__ attn_bf,     // [B,T,S] bf16 for context GEMM
    int B, int T, int S, int D) {
    __shared__ float qrow[512];
    __shared__ float sscale[512];
    __shared__ float sscores[512];
    __shared__ float red[256];

    const int tid  = threadIdx.x;
    const int lane = tid & 31;
    const int wave = tid >> 5;
    const int row  = blockIdx.x;              // b*T + t
    const int b    = row / T;

    const float* qp = q + (size_t)row * D;
    for (int d = tid; d < D; d += 256) { qrow[d] = qp[d]; sscale[d] = scale[d]; }
    __syncthreads();

    for (int s = wave; s < S; s += 8) {
        const float* kp = k + ((size_t)b * S + s) * D;
        if (s + 8 < S) __builtin_prefetch(kp + 8 * (size_t)D, 0, 0);  // global_prefetch_b8
        float part = 0.0f;
        for (int d = lane * 4; d < D; d += 128) {
            float4 kv = *(const float4*)(kp + d);
            part += sscale[d + 0] * DEV_TANH(qrow[d + 0] + kv.x);
            part += sscale[d + 1] * DEV_TANH(qrow[d + 1] + kv.y);
            part += sscale[d + 2] * DEV_TANH(qrow[d + 2] + kv.z);
            part += sscale[d + 3] * DEV_TANH(qrow[d + 3] + kv.w);
        }
        for (int off = 16; off > 0; off >>= 1)
            part += __shfl_xor(part, off, 32);
        if (lane == 0)
            sscores[s] = mask[(size_t)b * S + s] ? part : -1e9f;
    }
    __syncthreads();

    // rowmax
    float m = -3.4e38f;
    for (int s = tid; s < S; s += 256) m = fmaxf(m, sscores[s]);
    red[tid] = m; __syncthreads();
    for (int off = 128; off > 0; off >>= 1) {
        if (tid < off) red[tid] = fmaxf(red[tid], red[tid + off]);
        __syncthreads();
    }
    const float rowmax = red[0];
    __syncthreads();

    // exp + sum
    float sum = 0.0f;
    for (int s = tid; s < S; s += 256) {
        float e = __expf(sscores[s] - rowmax);
        sscores[s] = e;
        sum += e;
    }
    red[tid] = sum; __syncthreads();
    for (int off = 128; off > 0; off >>= 1) {
        if (tid < off) red[tid] += red[tid + off];
        __syncthreads();
    }
    const float inv = 1.0f / red[0];
    __syncthreads();

    for (int s = tid; s < S; s += 256) {
        float w = sscores[s] * inv;
        attn_out[(size_t)row * S + s] = w;
        attn_bf[(size_t)row * S + s] = f32_to_bf16(w);
    }
}

// ---------------------------------------------------------------------------
extern "C" void kernel_launch(void* const* d_in, const int* in_sizes, int n_in,
                              void* d_out, int out_size, void* d_ws, size_t ws_size,
                              hipStream_t stream) {
    (void)in_sizes; (void)n_in; (void)out_size; (void)ws_size;

    const float*         query = (const float*)d_in[0];         // [8,64,512]
    const float*         value = (const float*)d_in[1];         // [8,512,512]
    const unsigned char* mask  = (const unsigned char*)d_in[2]; // [8,512] bool
    const float*         W1w   = (const float*)d_in[3];         // [512,512]
    const float*         W1b   = (const float*)d_in[4];         // [512]
    const float*         W2w   = (const float*)d_in[5];         // [512,512]
    const float*         W2b   = (const float*)d_in[6];         // [512]
    const float*         scale = (const float*)d_in[7];         // [512]

    const int Bz = 8, T = 64, S = 512, D = 512;

    float* ctx_out  = (float*)d_out;                      // [B,T,D]
    float* attn_out = (float*)d_out + (size_t)Bz * T * D; // [B,T,S]

    // workspace carve-up (256B aligned)
    char* w = (char*)d_ws;
    auto alloc = [&](size_t bytes) {
        char* p = w;
        w += (bytes + 255) & ~(size_t)255;
        return p;
    };
    float*          q_f      = (float*)alloc(sizeof(float) * Bz * T * D);
    float*          k_f      = (float*)alloc(sizeof(float) * Bz * S * D);
    unsigned short* query_bf = (unsigned short*)alloc(2ull * Bz * T * D);
    unsigned short* value_bf = (unsigned short*)alloc(2ull * Bz * S * D);
    unsigned short* W1_bf    = (unsigned short*)alloc(2ull * D * D);
    unsigned short* W2_bf    = (unsigned short*)alloc(2ull * D * D);
    unsigned short* attn_bf  = (unsigned short*)alloc(2ull * Bz * T * S);

    auto cvt = [&](const float* in, unsigned short* out, int n) {
        cvt_bf16_kernel<<<(n + 255) / 256, 256, 0, stream>>>(in, out, n);
    };
    cvt(query, query_bf, Bz * T * D);
    cvt(value, value_bf, Bz * S * D);
    cvt(W1w, W1_bf, D * D);
    cvt(W2w, W2_bf, D * D);

    // q = query @ W1 + W1_b     (M=64, N=512, K=512, batch 8)
    gemm_bf16_wmma<<<dim3(D / 128, T / 64, Bz), 256, 0, stream>>>(
        query_bf, W1_bf, W1b, q_f, T, D, D, (long)T * D, 0L, (long)T * D);

    // k = value @ W2 + W2_b     (M=512, N=512, K=512, batch 8)
    gemm_bf16_wmma<<<dim3(D / 128, S / 64, Bz), 256, 0, stream>>>(
        value_bf, W2_bf, W2b, k_f, S, D, D, (long)S * D, 0L, (long)S * D);

    // scores + mask + softmax   (one block per (b,t))
    score_softmax_kernel<<<dim3(Bz * T), 256, 0, stream>>>(
        q_f, k_f, scale, mask, attn_out, attn_bf, Bz, T, S, D);

    // context = attn @ value    (M=64, N=512, K=512, batch 8)
    gemm_bf16_wmma<<<dim3(D / 128, T / 64, Bz), 256, 0, stream>>>(
        attn_bf, value_bf, nullptr, ctx_out, T, D, S,
        (long)T * S, (long)S * D, (long)T * D);
}
